// VectorQuantizer_72516227826204
// MI455X (gfx1250) — compile-verified, matrix operand order checked by reading, not run on hardware
//
#include <hip/hip_runtime.h>
#include <hip/hip_bf16.h>

typedef __attribute__((ext_vector_type(16))) __bf16 v16bf;
typedef __attribute__((ext_vector_type(8)))  float  v8f;
typedef __attribute__((ext_vector_type(4)))  int    v4i;

// Problem constants (from reference): z (8,256,8,32,32) f32, embed (1024,256) f32
#define NB 8
#define NC 256          // code_dim / channels
#define SPATIAL 8192    // 8*32*32
#define NROWS 65536     // NB*SPATIAL
#define NCODES 1024
#define ZBATCH 2097152  // NC*SPATIAL

#define ZQ_OFF   0
#define IDX_OFF  16777216
#define LOSS_OFF 16842752
#define PERP_OFF 16842753

#if __has_builtin(__builtin_amdgcn_global_load_async_to_lds_b128)
#define HAVE_ASYNC_LDS 1
#else
#define HAVE_ASYNC_LDS 0
#endif

// AS1 (global) pointer via integer round-trip (AS0/AS1 share representation);
// AS3 (LDS) pointer via explicit addrspacecast.
#define GV4I(p) ((__attribute__((address_space(1))) v4i*)(unsigned long long)(p))
#define LV4I(p) ((__attribute__((address_space(3))) v4i*)(p))

__device__ inline void waitAsyncLds() {
#if HAVE_ASYNC_LDS
#if __has_builtin(__builtin_amdgcn_s_wait_asynccnt)
  __builtin_amdgcn_s_wait_asynccnt(0);
#else
  asm volatile("s_wait_asynccnt 0" ::: "memory");
#endif
#endif
}

__device__ inline float blockReduceSum(float v, float* sbuf, int nw) {
  int lane = threadIdx.x & 31;
  int w = threadIdx.x >> 5;
#pragma unroll
  for (int m = 16; m; m >>= 1) v += __shfl_xor(v, m, 32);
  if (lane == 0) sbuf[w] = v;
  __syncthreads();
  float r = (lane < nw) ? sbuf[lane] : 0.0f;
#pragma unroll
  for (int m = 16; m; m >>= 1) r += __shfl_xor(r, m, 32);
  __syncthreads();
  return r;  // full sum valid in every thread
}

// ---------------------------------------------------------------------------
// Kernel 1: convert codebook to bf16, compute ||e||^2 per code, zero counters
// ---------------------------------------------------------------------------
__global__ __launch_bounds__(256) void vq_prep_kernel(
    const float* __restrict__ embed, __bf16* __restrict__ ebf,
    float* __restrict__ esq, int* __restrict__ counts,
    float* __restrict__ partials) {
  __shared__ float sbuf[8];
  const int code = blockIdx.x;
  const int t = threadIdx.x;
  float v = embed[code * NC + t];
  ebf[code * NC + t] = (__bf16)v;
  float s = blockReduceSum(v * v, sbuf, 8);
  if (t == 0) {
    esq[code] = s;
    counts[code] = 0;
    if (code < 512) partials[code] = 0.0f;
  }
}

// ---------------------------------------------------------------------------
// Kernel 2: distances via WMMA bf16 + argmin + fused commitment loss + hist
// 512 blocks x 256 threads (8 waves). Wave w handles rows [tile*16, tile*16+16)
// Codebook staged 128 codes (64KB bf16) at a time into LDS, double-buffered
// with async global->LDS loads, shared by all 8 waves.
// ---------------------------------------------------------------------------
__global__ __launch_bounds__(256) void vq_argmin_kernel(
    const float* __restrict__ z, const __bf16* __restrict__ ebf,
    const float* __restrict__ esq, int* __restrict__ idx_out,
    int* __restrict__ counts, float* __restrict__ partials) {
  __shared__ __align__(16) __bf16 sE[2][128 * NC];  // 2 x 64 KB stages
  __shared__ float sS[2][128];
  __shared__ float sbuf[8];

  const int wave = threadIdx.x >> 5;
  const int lane = threadIdx.x & 31;
  const int m = lane & 15;    // row within tile (A) / column within subtile (B)
  const int kh = lane >> 4;   // K-half select per CDNA5 16-bit WMMA layout
  const int tile = blockIdx.x * 8 + wave;
  const int n = tile * 16 + m;                 // global flat row this lane loads
  const long b_ = (long)(n >> 13);
  const long off = (long)(n & 8191);
  const float* zrow = z + b_ * ZBATCH + off;   // element c at zrow[c*8192]

  // async (or fallback) copy of one 128-code stage into LDS buffer `buf`
  auto stageLoad = [&](int s, int buf) {
    const uint4* src = (const uint4*)(ebf + (size_t)s * 128 * NC) + threadIdx.x;
    uint4* dst = (uint4*)(&sE[buf][0]) + threadIdx.x;
#if HAVE_ASYNC_LDS
#pragma unroll
    for (int i = 0; i < 16; ++i)
      __builtin_amdgcn_global_load_async_to_lds_b128(
          GV4I(src + 256 * i), LV4I(dst + 256 * i), 0, 0);
#else
#pragma unroll
    for (int i = 0; i < 16; ++i) dst[256 * i] = src[256 * i];
#endif
    if (threadIdx.x < 128) sS[buf][threadIdx.x] = esq[s * 128 + threadIdx.x];
  };

  // --- load A fragments (16x32 bf16, documented VGPR layout), fp32 ||z||^2 ---
  v16bf af[8];
  float zsq = 0.0f;
#pragma unroll
  for (int kc = 0; kc < 8; ++kc) {
#pragma unroll
    for (int v = 0; v < 8; ++v) {
      const int kp = kc * 32 + ((v & 4) ? 16 : 0) + kh * 8 + (v & 3) * 2;
      float f0 = zrow[(long)kp * SPATIAL];
      float f1 = zrow[(long)(kp + 1) * SPATIAL];
      zsq += f0 * f0 + f1 * f1;
      af[kc][2 * v]     = (__bf16)f0;
      af[kc][2 * v + 1] = (__bf16)f1;
    }
  }

  float mind[8];
  int   mini[8];
#pragma unroll
  for (int r = 0; r < 8; ++r) { mind[r] = 3.4e38f; mini[r] = 0; }

  // prologue: fetch stage 0
  stageLoad(0, 0);
  waitAsyncLds();
  __syncthreads();

  for (int stage = 0; stage < 8; ++stage) {
    const int buf = stage & 1;
    if (stage < 7) stageLoad(stage + 1, buf ^ 1);  // overlap with compute below
    if (stage < 6)  // hint stage+2 toward L2 (global_prefetch_b8)
      __builtin_prefetch(ebf + (size_t)(stage + 2) * 128 * NC + threadIdx.x * 128, 0, 0);

#pragma unroll
    for (int j = 0; j < 8; ++j) {   // 16-code subtiles within the stage
      const int col = j * 16 + m;
      v8f acc;
#pragma unroll
      for (int r = 0; r < 8; ++r) acc[r] = 0.0f;
#pragma unroll
      for (int kc = 0; kc < 8; ++kc) {
        union { v16bf v; uint4 q[2]; } bf;
        const uint4* bp = (const uint4*)(&sE[buf][0] + col * NC + kc * 32 + kh * 8);
        bf.q[0] = bp[0];   // K = kc*32 + kh*8 .. +7
        bf.q[1] = bp[2];   // K = kc*32 + 16 + kh*8 .. +7
        acc = __builtin_amdgcn_wmma_f32_16x16x32_bf16(
            false, af[kc], false, bf.v, (short)0, acc, false, false);
      }
      const float es = sS[buf][j * 16 + m];
      const int code = stage * 128 + j * 16 + m;
#pragma unroll
      for (int r = 0; r < 8; ++r) {
        float d = es - 2.0f * acc[r];
        if (d < mind[r]) { mind[r] = d; mini[r] = code; }
      }
    }

    if (stage < 7) {
      waitAsyncLds();   // next buffer fully landed in LDS
      __syncthreads();  // all waves done reading `buf`, next stage visible
    }
  }

  // --- reduce argmin across the 16 lanes of each D half (first-index ties) ---
#pragma unroll
  for (int r = 0; r < 8; ++r) {
#pragma unroll
    for (int mask = 8; mask >= 1; mask >>= 1) {
      float od = __shfl_xor(mind[r], mask, 32);
      int   oi = __shfl_xor(mini[r], mask, 32);
      if (od < mind[r] || (od == mind[r] && oi < mini[r])) {
        mind[r] = od; mini[r] = oi;
      }
    }
  }

  // --- fused loss: ||z - e||^2 = ||z||^2 + (||e||^2 - 2 z.e)_min ---
  float zfull = zsq + __shfl_xor(zsq, 16, 32);       // full-row ||z||^2 at lane m
  float zg = ((m >> 3) == kh) ? zfull : 0.0f;        // rows owned by this D half
#pragma unroll
  for (int mask = 8; mask >= 1; mask >>= 1) zg += __shfl_xor(zg, mask, 32);

  float lossLocal = 0.0f;
  if (m == 0) {
    const int row0 = tile * 16;
    float s = zg;
#pragma unroll
    for (int r = 0; r < 8; ++r) {
      s += mind[r];
      const int row = row0 + r + 8 * kh;
      idx_out[row] = mini[r];
      atomicAdd(&counts[mini[r]], 1);
    }
    lossLocal = s;
  }
  __syncthreads();
  float bs = blockReduceSum(lossLocal, sbuf, 8);
  if (threadIdx.x == 0) partials[blockIdx.x] = bs;
}

// ---------------------------------------------------------------------------
// Kernel 3: z_q gather + layout transpose back to (B,C,D,H,W); indices->float
// 1024 blocks x 256 threads; block handles 64 consecutive rows. Coalesced
// stores over the spatial index; embed gathers served from L2.
// ---------------------------------------------------------------------------
__global__ __launch_bounds__(256) void vq_gather_kernel(
    const float* __restrict__ embed, const int* __restrict__ idx,
    float* __restrict__ out) {
  const int n0 = blockIdx.x * 64;
  const int tn = threadIdx.x & 63;
  const int tc = threadIdx.x >> 6;
  const int n = n0 + tn;
  const long b_ = (long)(n >> 13);
  const long off = (long)(n & 8191);
  const long base = b_ * ZBATCH + off;
  const int code = idx[n];
  const float* erow = embed + (long)code * NC;
  float* zq = out + ZQ_OFF;
#pragma unroll 8
  for (int ci = 0; ci < 64; ++ci) {
    const int c = ci * 4 + tc;
    zq[base + (long)c * SPATIAL] = erow[c];
  }
  if (threadIdx.x < 64)
    out[IDX_OFF + n0 + threadIdx.x] = (float)idx[n0 + threadIdx.x];
}

// ---------------------------------------------------------------------------
// Kernel 4: deterministic final reductions -> loss, perplexity
// ---------------------------------------------------------------------------
__global__ __launch_bounds__(1024) void vq_finalize_kernel(
    const float* __restrict__ partials, const int* __restrict__ counts,
    float* __restrict__ out) {
  __shared__ float sbuf[32];
  const int t = threadIdx.x;
  float pl = (t < 512) ? partials[t] : 0.0f;
  float p = (float)counts[t] * (1.0f / 65536.0f);
  float ent = p * logf(p + 1e-10f);
  float sl = blockReduceSum(pl, sbuf, 32);
  float se = blockReduceSum(ent, sbuf, 32);
  if (t == 0) {
    out[LOSS_OFF] = 0.25f * sl / 16777216.0f;  // BETA * mean
    out[PERP_OFF] = expf(-se);
  }
}

extern "C" void kernel_launch(void* const* d_in, const int* in_sizes, int n_in,
                              void* d_out, int out_size, void* d_ws, size_t ws_size,
                              hipStream_t stream) {
  const float* z     = (const float*)d_in[0];
  const float* embed = (const float*)d_in[1];
  float* out = (float*)d_out;

  char* ws = (char*)d_ws;
  __bf16* ebf     = (__bf16*)(ws);             // 1024*256*2 = 524288 B
  float*  esq     = (float*)(ws + 524288);     // 4096 B
  int*    idx     = (int*)(ws + 528384);       // 65536*4 = 262144 B
  int*    counts  = (int*)(ws + 790528);       // 4096 B
  float*  partials= (float*)(ws + 794624);     // 2048 B (512 used)

  vq_prep_kernel<<<NCODES, 256, 0, stream>>>(embed, ebf, esq, counts, partials);
  vq_argmin_kernel<<<512, 256, 0, stream>>>(z, ebf, esq, idx, counts, partials);
  vq_gather_kernel<<<1024, 256, 0, stream>>>(embed, idx, out);
  vq_finalize_kernel<<<1, 1024, 0, stream>>>(partials, counts, out);
}
